// SelfAttention_40750649704667
// MI455X (gfx1250) — matrix operand order repaired and checked by reading, not compiled, not run on hardware
//
#include <hip/hip_runtime.h>
#include <hip/hip_bf16.h>

// ---------------------------------------------------------------------------
// Self-attention forward, B=8 T=1024 D=768 H=12 HD=64, bf16 WMMA + TDM (wave32)
// ---------------------------------------------------------------------------

typedef __attribute__((ext_vector_type(16))) __bf16 bf16x16;
typedef __attribute__((ext_vector_type(8)))  __bf16 bf16x8;
typedef __attribute__((ext_vector_type(8)))  float  f32x8;
typedef __attribute__((ext_vector_type(4)))  unsigned int uint32x4;
typedef __attribute__((ext_vector_type(4)))  int int32x4;
typedef __attribute__((ext_vector_type(8)))  int int32x8;

static constexpr int Bn  = 8;
static constexpr int Tn  = 1024;
static constexpr int Dn  = 768;
static constexpr int Hn  = 12;
static constexpr int HDn = 64;
static constexpr int BHn = Bn * Hn;      // 96
static constexpr int BTn = Bn * Tn;      // 8192

__device__ __forceinline__ bf16x16 load_frag_split(const __bf16* p0, const __bf16* p1) {
  union { bf16x16 v; bf16x8 h[2]; } u;
  u.h[0] = *(const bf16x8*)p0;   // elements 0..7
  u.h[1] = *(const bf16x8*)p1;   // elements 8..15
  return u.v;
}
__device__ __forceinline__ bf16x16 load_frag_contig(const __bf16* p) {
  union { bf16x16 v; bf16x8 h[2]; } u;
  u.h[0] = *(const bf16x8*)p;
  u.h[1] = *(const bf16x8*)(p + 8);
  return u.v;
}
__device__ __forceinline__ f32x8 wmma_bf16(bf16x16 a, bf16x16 b, f32x8 c) {
  return __builtin_amdgcn_wmma_f32_16x16x32_bf16(false, a, false, b, (short)0, c,
                                                 false, false);
}

// ---------------------------------------------------------------------------
// Tensor Data Mover: 2-D tile load (global -> LDS), data_size = 2 bytes.
// Descriptor packing per CDNA5 ISA ch.8 (group0 128b, group1 256b, groups 2/3
// zero => tile_dim2/3 unused).  Issued by one wave; tracked by TENSORcnt.
// This toolchain exposes the 6-arg builtin form:
//   (uint32x4 g0, int32x8 g1, int32x4 g2, int32x4 g3, int32x8 gx, i32 cpol)
// ---------------------------------------------------------------------------
__device__ __forceinline__ unsigned rfl(unsigned v) {
  return (unsigned)__builtin_amdgcn_readfirstlane((int)v);
}
__device__ __forceinline__ void tdm_load_2d(unsigned lds_off, const void* gptr,
                                            unsigned tensor_d0, unsigned tensor_d1,
                                            unsigned tile_d0, unsigned tile_d1,
                                            unsigned stride0_elems) {
  const unsigned long long ga = (unsigned long long)gptr;
  uint32x4 g0;
  g0[0] = rfl(1u);                                   // count=1, user mode
  g0[1] = rfl(lds_off);                              // lds_addr (bytes)
  g0[2] = rfl((unsigned)ga);                         // global_addr[31:0]
  g0[3] = rfl(((unsigned)(ga >> 32) & 0x01ffffffu) | (2u << 30)); // [56:32]|type=2
  int32x8 g1;
  g1[0] = (int)rfl(1u << 16);                        // data_size=1 -> 2 bytes
  g1[1] = (int)rfl((tensor_d0 & 0xffffu) << 16);     // tensor_dim0[15:0]
  g1[2] = (int)rfl(((tensor_d0 >> 16) & 0xffffu) | ((tensor_d1 & 0xffffu) << 16));
  g1[3] = (int)rfl(((tensor_d1 >> 16) & 0xffffu) | ((tile_d0 & 0xffffu) << 16));
  g1[4] = (int)rfl(tile_d1 & 0xffffu);               // tile_dim1 (tile_dim2=0)
  g1[5] = (int)rfl(stride0_elems);                   // tensor_dim0_stride[31:0]
  g1[6] = 0;
  g1[7] = 0;
  int32x4 z4 = {0, 0, 0, 0};
  int32x8 z8 = {0, 0, 0, 0, 0, 0, 0, 0};
  __builtin_amdgcn_tensor_load_to_lds(g0, g1, z4, z4, z8, 0);
}

// ---------------------------------------------------------------------------
// fp32 -> bf16 conversion / transpose kernels
// ---------------------------------------------------------------------------
__global__ void cvt_x_kernel(const float* __restrict__ x, __bf16* __restrict__ xb, int n) {
  for (int i = blockIdx.x * blockDim.x + threadIdx.x; i < n; i += gridDim.x * blockDim.x)
    xb[i] = (__bf16)x[i];
}

// WT[n*K + k] = W[k*N + n]  (write side contiguous)
__global__ void transpose_cvt_kernel(const float* __restrict__ W, __bf16* __restrict__ WT,
                                     int K, int N) {
  int total = K * N;
  for (int i = blockIdx.x * blockDim.x + threadIdx.x; i < total; i += gridDim.x * blockDim.x) {
    int n = i / K, k = i - n * K;
    WT[i] = (__bf16)W[(size_t)k * N + n];
  }
}

// ---------------------------------------------------------------------------
// GEMM core: 128 threads (4 waves), block tile 64(M) x 256(N), K = 768.
// A tile (64x32 bf16) double-buffered in LDS via TDM; B fragments from the
// pre-transposed weight matrix (contiguous per lane).  16 WMMAs/wave/k-step.
// acc[mi][nt] covers rows mi*16.., cols (wave*64 + nt*16)..
// ---------------------------------------------------------------------------
struct AccTile { f32x8 c[4][4]; };

__device__ __forceinline__ void gemm_64x256(const __bf16* __restrict__ Agl /*row m0, col 0*/,
                                            const __bf16* __restrict__ BtCol /*col block base*/,
                                            __bf16 (*Abuf)[64 * 32], AccTile& t) {
  const int tid  = threadIdx.x;
  const int lane = tid & 31;
  const int hi   = (lane >= 16);
  const int l15  = lane & 15;

  const __bf16* bptr = BtCol + (size_t)l15 * Dn + (hi ? 16 : 0);

  const int nks = Dn / 32;  // 24
  if (tid < 32) {
    tdm_load_2d((unsigned)(unsigned long long)(void*)&Abuf[0][0], Agl,
                Dn, BTn, 32, 64, Dn);
    __builtin_amdgcn_s_wait_tensorcnt(0);
  }
  __syncthreads();

  for (int ks = 0; ks < nks; ++ks) {
    const int cur = ks & 1;
    if ((tid < 32) && (ks + 1 < nks))
      tdm_load_2d((unsigned)(unsigned long long)(void*)&Abuf[cur ^ 1][0],
                  Agl + (size_t)(ks + 1) * 32, Dn, BTn, 32, 64, Dn);

    bf16x16 a[4];
#pragma unroll
    for (int mi = 0; mi < 4; ++mi) {
      const __bf16* ar = &Abuf[cur][(mi * 16 + l15) * 32 + (hi ? 16 : 0)];
      a[mi] = load_frag_contig(ar);
    }
    const int kk = ks * 32;
#pragma unroll
    for (int nt = 0; nt < 4; ++nt) {
      bf16x16 b = load_frag_contig(bptr + (size_t)nt * 16 * Dn + kk);
#pragma unroll
      for (int mi = 0; mi < 4; ++mi)
        t.c[mi][nt] = wmma_bf16(a[mi], b, t.c[mi][nt]);
    }

    if ((tid < 32) && (ks + 1 < nks))
      __builtin_amdgcn_s_wait_tensorcnt(0);
    __syncthreads();
  }
}

// ---------------------------------------------------------------------------
// QKV GEMM: xb[8192,768] @ WqkvT[2304,768]^T -> Q/K (bf16 [BH,T,64]),
// V transposed (bf16 [BH,64,T]).  grid: (BT/64, 2304/256) = (128, 9), block 128.
// ---------------------------------------------------------------------------
__global__ __launch_bounds__(128)
void qkv_gemm_kernel(const __bf16* __restrict__ xb, const __bf16* __restrict__ wT,
                     const float* __restrict__ bias,
                     __bf16* __restrict__ Q, __bf16* __restrict__ K,
                     __bf16* __restrict__ VT) {
  __shared__ __align__(16) __bf16 Abuf[2][64 * 32];
  const int tid  = threadIdx.x;
  const int lane = tid & 31;
  const int wave = tid >> 5;
  const int hi   = (lane >= 16);
  const int l15  = lane & 15;
  const int m0   = blockIdx.x * 64;
  const int n0   = blockIdx.y * 256 + wave * 64;

  AccTile t = {};
  gemm_64x256(xb + (size_t)m0 * Dn, wT + (size_t)n0 * Dn, Abuf, t);

#pragma unroll
  for (int mi = 0; mi < 4; ++mi) {
#pragma unroll
    for (int nt = 0; nt < 4; ++nt) {
#pragma unroll
      for (int i = 0; i < 8; ++i) {
        const int bt = m0 + mi * 16 + i + (hi ? 8 : 0);   // b*T + t
        const int n  = n0 + nt * 16 + l15;                // 0..2303
        const float v = t.c[mi][nt][i] + bias[n];
        const int bb = bt >> 10, tt = bt & 1023;
        const int sect = n / Dn, nn = n % Dn;
        const int h = nn >> 6, hd = nn & 63;
        const int bh = bb * Hn + h;
        const __bf16 bv = (__bf16)v;
        if (sect == 0)      Q[((size_t)bh * Tn + tt) * HDn + hd] = bv;
        else if (sect == 1) K[((size_t)bh * Tn + tt) * HDn + hd] = bv;
        else                VT[((size_t)bh * HDn + hd) * Tn + tt] = bv;
      }
    }
  }
}

// ---------------------------------------------------------------------------
// Proj GEMM: AO[8192,768] @ WprojT[768,768]^T + bias -> out fp32.
// grid: (128, 3), block 128.
// ---------------------------------------------------------------------------
__global__ __launch_bounds__(128)
void proj_gemm_kernel(const __bf16* __restrict__ A, const __bf16* __restrict__ wT,
                      const float* __restrict__ bias, float* __restrict__ out) {
  __shared__ __align__(16) __bf16 Abuf[2][64 * 32];
  const int tid  = threadIdx.x;
  const int lane = tid & 31;
  const int wave = tid >> 5;
  const int hi   = (lane >= 16);
  const int l15  = lane & 15;
  const int m0   = blockIdx.x * 64;
  const int n0   = blockIdx.y * 256 + wave * 64;

  AccTile t = {};
  gemm_64x256(A + (size_t)m0 * Dn, wT + (size_t)n0 * Dn, Abuf, t);

#pragma unroll
  for (int mi = 0; mi < 4; ++mi) {
#pragma unroll
    for (int nt = 0; nt < 4; ++nt) {
#pragma unroll
      for (int i = 0; i < 8; ++i) {
        const int bt = m0 + mi * 16 + i + (hi ? 8 : 0);
        const int n  = n0 + nt * 16 + l15;
        out[(size_t)bt * Dn + n] = t.c[mi][nt][i] + bias[n];
      }
    }
  }
}

// ---------------------------------------------------------------------------
// Flash attention: one wave per (bh, 16-query tile); stream 32-key blocks.
// grid: (T/16, BH) = (64, 96), block: 32
// ---------------------------------------------------------------------------
__global__ __launch_bounds__(32)
void attn_kernel(const __bf16* __restrict__ Q, const __bf16* __restrict__ Kb,
                 const __bf16* __restrict__ VT, __bf16* __restrict__ AO) {
  __shared__ __align__(16) __bf16 Plds[16 * 32];

  const int lane = threadIdx.x;
  const int hi   = lane >= 16;
  const int l15  = lane & 15;
  const int qbase = blockIdx.x * 16;
  const int bh    = blockIdx.y;

  // Q A-fragments (d = 0..31 and 32..63)
  const __bf16* qrow = Q + ((size_t)bh * Tn + (qbase + l15)) * HDn + (hi ? 16 : 0);
  const bf16x16 aq0 = load_frag_contig(qrow);
  const bf16x16 aq1 = load_frag_contig(qrow + 32);

  f32x8 o0 = {}, o1 = {}, o2 = {}, o3 = {};
  float mrun[8], lrun[8];
#pragma unroll
  for (int i = 0; i < 8; ++i) { mrun[i] = -1.0e30f; lrun[i] = 0.0f; }

  const int jmax = (qbase + 15) / 32;
  for (int jb = 0; jb <= jmax; ++jb) {
    const int kbase = jb * 32;

    // ---- S = Q @ K^T  (16 queries x 32 keys) ----
    f32x8 s[2];
#pragma unroll
    for (int f = 0; f < 2; ++f) {
      const __bf16* krow =
          Kb + ((size_t)bh * Tn + (kbase + f * 16 + l15)) * HDn + (hi ? 16 : 0);
      f32x8 c = {};
      c = wmma_bf16(aq0, load_frag_contig(krow), c);        // d 0..31
      c = wmma_bf16(aq1, load_frag_contig(krow + 32), c);   // d 32..63
      s[f] = c;
    }

    // ---- scale + causal mask ----
#pragma unroll
    for (int f = 0; f < 2; ++f)
#pragma unroll
      for (int i = 0; i < 8; ++i) {
        const int kg = kbase + f * 16 + l15;
        const int qg = qbase + i + (hi ? 8 : 0);
        const float v = s[f][i] * 0.125f;           // 1/sqrt(64)
        s[f][i] = (kg > qg) ? -1.0e30f : v;
      }

    // ---- online softmax (row reductions across 16 lanes of each row) ----
#pragma unroll
    for (int i = 0; i < 8; ++i) {
      float rm = fmaxf(s[0][i], s[1][i]);
      rm = fmaxf(rm, __shfl_xor(rm, 1));
      rm = fmaxf(rm, __shfl_xor(rm, 2));
      rm = fmaxf(rm, __shfl_xor(rm, 4));
      rm = fmaxf(rm, __shfl_xor(rm, 8));
      const float mnew  = fmaxf(mrun[i], rm);
      const float alpha = __expf(mrun[i] - mnew);
      const float p0 = __expf(s[0][i] - mnew);
      const float p1 = __expf(s[1][i] - mnew);
      s[0][i] = p0; s[1][i] = p1;
      float rs = p0 + p1;
      rs += __shfl_xor(rs, 1);
      rs += __shfl_xor(rs, 2);
      rs += __shfl_xor(rs, 4);
      rs += __shfl_xor(rs, 8);
      lrun[i] = lrun[i] * alpha + rs;
      mrun[i] = mnew;
      o0[i] *= alpha; o1[i] *= alpha; o2[i] *= alpha; o3[i] *= alpha;
    }

    // ---- P: C-layout -> A-layout via LDS round trip ----
#pragma unroll
    for (int f = 0; f < 2; ++f)
#pragma unroll
      for (int i = 0; i < 8; ++i) {
        const int r = i + (hi ? 8 : 0);
        const int c = f * 16 + l15;
        Plds[r * 32 + c] = (__bf16)s[f][i];
      }
    __syncthreads();
    const __bf16* pr = Plds + l15 * 32 + (hi ? 16 : 0);
    const bf16x16 ap = load_frag_contig(pr);
    __syncthreads();

    // ---- O += P @ V  (B-fragments contiguous thanks to VT layout) ----
    const __bf16* vbase = VT + (size_t)bh * HDn * Tn + kbase + (hi ? 16 : 0);
    o0 = wmma_bf16(ap, load_frag_contig(vbase + (size_t)(0 * 16 + l15) * Tn), o0);
    o1 = wmma_bf16(ap, load_frag_contig(vbase + (size_t)(1 * 16 + l15) * Tn), o1);
    o2 = wmma_bf16(ap, load_frag_contig(vbase + (size_t)(2 * 16 + l15) * Tn), o2);
    o3 = wmma_bf16(ap, load_frag_contig(vbase + (size_t)(3 * 16 + l15) * Tn), o3);
  }

  // ---- epilogue: O /= l, store as [B, T, H*HD] bf16 ----
  const int bb = bh / Hn, h = bh % Hn;
#pragma unroll
  for (int i = 0; i < 8; ++i) {
    const float inv = 1.0f / lrun[i];
    const int qg = qbase + i + (hi ? 8 : 0);
    const size_t base = (((size_t)bb * Tn + qg) * Hn + h) * HDn + l15;
    AO[base + 0]  = (__bf16)(o0[i] * inv);
    AO[base + 16] = (__bf16)(o1[i] * inv);
    AO[base + 32] = (__bf16)(o2[i] * inv);
    AO[base + 48] = (__bf16)(o3[i] * inv);
  }
}

// ---------------------------------------------------------------------------
// Launch
// ---------------------------------------------------------------------------
extern "C" void kernel_launch(void* const* d_in, const int* in_sizes, int n_in,
                              void* d_out, int out_size, void* d_ws, size_t ws_size,
                              hipStream_t stream) {
  const float* x     = (const float*)d_in[0];
  const float* Wqkv  = (const float*)d_in[1];
  const float* bqkv  = (const float*)d_in[2];
  const float* Wproj = (const float*)d_in[3];
  const float* bproj = (const float*)d_in[4];
  float* out = (float*)d_out;

  char* ws = (char*)d_ws;
  size_t off = 0;
  auto alloc = [&](size_t bytes) -> void* {
    void* p = ws + off;
    off += (bytes + 255) & ~(size_t)255;
    return p;
  };
  __bf16* xb     = (__bf16*)alloc((size_t)BTn * Dn * 2);
  __bf16* wqkvT  = (__bf16*)alloc((size_t)3 * Dn * Dn * 2);
  __bf16* wprojT = (__bf16*)alloc((size_t)Dn * Dn * 2);
  __bf16* Qb     = (__bf16*)alloc((size_t)BHn * Tn * HDn * 2);
  __bf16* Kbuf   = (__bf16*)alloc((size_t)BHn * Tn * HDn * 2);
  __bf16* VTb    = (__bf16*)alloc((size_t)BHn * HDn * Tn * 2);
  __bf16* AO     = (__bf16*)alloc((size_t)BTn * Dn * 2);

  cvt_x_kernel<<<1024, 256, 0, stream>>>(x, xb, BTn * Dn);
  transpose_cvt_kernel<<<1024, 256, 0, stream>>>(Wqkv, wqkvT, Dn, 3 * Dn);
  transpose_cvt_kernel<<<512, 256, 0, stream>>>(Wproj, wprojT, Dn, Dn);

  qkv_gemm_kernel<<<dim3(BTn / 64, (3 * Dn) / 256), 128, 0, stream>>>(
      xb, wqkvT, bqkv, Qb, Kbuf, VTb);

  attn_kernel<<<dim3(Tn / 16, BHn), 32, 0, stream>>>(Qb, Kbuf, VTb, AO);

  proj_gemm_kernel<<<dim3(BTn / 64, Dn / 256), 128, 0, stream>>>(AO, wprojT, bproj, out);
}
